// ScaledMultiHeadAttention_30313879175869
// MI455X (gfx1250) — compile-verified, matrix operand order checked by reading, not run on hardware
//
#include <hip/hip_runtime.h>

// ---------------------------------------------------------------------------
// Multi-head attention, CDNA5 (gfx1250), wave32, WMMA f16->f32.
//   B=4, S=2048, D=1024, H=16, HD=64
// d_out = [ out (4*2048*1024 f32) | attn_weights (4*16*2048*2048 f32) ]
// ---------------------------------------------------------------------------

typedef __attribute__((ext_vector_type(16))) _Float16 v16h;
typedef __attribute__((ext_vector_type(8)))  _Float16 v8h;
typedef __attribute__((ext_vector_type(4)))  _Float16 v4h;
typedef __attribute__((ext_vector_type(8)))  float    v8f;

#define BB 4
#define SS 2048
#define DD 1024
#define HH 16
#define HD 64

__device__ __forceinline__ v8f wmma_f16(v16h a, v16h b, v8f c) {
  // D = A(16x32 f16) x B(32x16 f16) + C(16x16 f32)
  return __builtin_amdgcn_wmma_f32_16x16x32_f16(
      /*neg_a=*/false, a, /*neg_b=*/false, b,
      /*c_mod=*/(short)0, c, /*reuse_a=*/false, /*reuse_b=*/false);
}

// Async DMA one 16B chunk global -> LDS (GLOBAL_LOAD_ASYNC_TO_LDS_B128, GV
// mode). LDS address = low 32 bits of the generic shared pointer (wave-
// relative LDS byte offset); tracked with ASYNCcnt.
__device__ __forceinline__ void async_copy_b128(void* lds_dst, const void* gsrc) {
  unsigned lds = (unsigned)(size_t)lds_dst;
  unsigned long long ga = (unsigned long long)(size_t)gsrc;
  asm volatile("global_load_async_to_lds_b128 %0, %1, off"
               :: "v"(lds), "v"(ga) : "memory");
}
__device__ __forceinline__ void wait_async_all() {
  asm volatile("s_wait_asynccnt 0x0" ::: "memory");
}

// Load a 16x32 f16 fragment (A-operand layout; B-operand of shape 32x16 uses
// the same per-lane byte pattern with "row" = its N column) from a row-major
// array with row stride `rs` halves. Per ISA table: lanes 0-15 hold K=0..7 and
// K=16..23 (two contiguous 8-half groups); lanes 16-31 hold K=8..15 & 24..31.
__device__ __forceinline__ v16h ld_frag(const _Float16* base, int rs, int kbase, int lane) {
  const int i  = lane & 15;
  const int ko = (lane & 16) ? 8 : 0;
  const _Float16* p = base + (size_t)i * rs + kbase + ko;
  v8h lo = *(const v8h*)(p);
  v8h hi = *(const v8h*)(p + 16);
  v16h f;
#pragma unroll
  for (int t = 0; t < 8; ++t) { f[t] = lo[t]; f[t + 8] = hi[t]; }
  return f;
}

// ---------------------------------------------------------------------------
// f32 -> f16 convert (vectorized 4/thread)
// ---------------------------------------------------------------------------
__global__ __launch_bounds__(256) void cvt_f32_f16(const float* __restrict__ in,
                                                   _Float16* __restrict__ out, int n4) {
  int idx = blockIdx.x * blockDim.x + threadIdx.x;
  if (idx < n4) {
    float4 v = ((const float4*)in)[idx];
    v4h o = {(_Float16)v.x, (_Float16)v.y, (_Float16)v.z, (_Float16)v.w};
    *(v4h*)(out + (size_t)idx * 4) = o;
  }
}

// ---------------------------------------------------------------------------
// WMMA GEMM: C[M,N] = A[M,K](f16) * W[K,N](f16) + bias.
// Block tile 64(M) x 128(N), K-step 32, 8 waves (4 m-subtiles x 2 n-halves).
// A tile staged via async global->LDS DMA; B tile transposed via registers.
// MODE 0: write f32 C row-major (output projection).
// MODE 1: epilogue scatters qkv -> Q[B,H,S,HD], K[B,H,S,HD], V^T[B,H,HD,S] f16.
// ---------------------------------------------------------------------------
template <int MODE>
__global__ __launch_bounds__(256) void gemm16(const _Float16* __restrict__ A,
                                              const _Float16* __restrict__ W,
                                              const float* __restrict__ bias,
                                              float* __restrict__ Cout,
                                              _Float16* __restrict__ qh,
                                              _Float16* __restrict__ kh,
                                              _Float16* __restrict__ vTh,
                                              int M, int N, int K) {
  __shared__ __align__(16) _Float16 As[64 * 40];   // 64 rows x 32 k (+8 pad)
  __shared__ __align__(16) _Float16 Bs[128 * 40];  // transposed: [n][k] (+pad)

  const int tid  = threadIdx.x;
  const int lane = tid & 31;
  const int w    = tid >> 5;
  const int wm   = w & 3;    // m-subtile (16 rows)
  const int wn   = w >> 2;   // n-half (64 cols)
  const int m0   = blockIdx.y * 64;
  const int n0   = blockIdx.x * 128;

  v8f acc[4] = {};

  for (int k0 = 0; k0 < K; k0 += 32) {
    {  // A tile 64x32: one async 16B DMA per thread, global -> LDS
      int row = tid >> 2, ch = tid & 3;
      async_copy_b128(As + row * 40 + ch * 8,
                      A + (size_t)(m0 + row) * K + k0 + ch * 8);
    }
    {  // B tile 32x128, stored transposed so fragment k-groups are contiguous
      int k = tid >> 3, nch = tid & 7;
      const _Float16* src = W + (size_t)(k0 + k) * N + n0 + nch * 16;
      v8h b0 = *(const v8h*)(src);
      v8h b1 = *(const v8h*)(src + 8);
#pragma unroll
      for (int t = 0; t < 8; ++t) {
        Bs[(nch * 16 + t) * 40 + k]     = b0[t];
        Bs[(nch * 16 + 8 + t) * 40 + k] = b1[t];
      }
    }
    wait_async_all();
    __syncthreads();

    v16h af = ld_frag(As + wm * 16 * 40, 40, 0, lane);
#pragma unroll
    for (int j = 0; j < 4; ++j) {
      v16h bf = ld_frag(Bs + (wn * 64 + j * 16) * 40, 40, 0, lane);
      acc[j] = wmma_f16(af, bf, acc[j]);
    }
    __syncthreads();
  }

  // Epilogue. C layout: VGPR r -> lanes 0-15: M=r, N=lane; lanes 16-31: M=8+r.
  const int i  = lane & 15;
  const int mo = (lane & 16) ? 8 : 0;
#pragma unroll
  for (int j = 0; j < 4; ++j) {
    int n = n0 + wn * 64 + j * 16 + i;
    float bv = bias[n];
#pragma unroll
    for (int r = 0; r < 8; ++r) {
      int m = m0 + wm * 16 + mo + r;
      float val = acc[j][r] + bv;
      if (MODE == 0) {
        Cout[(size_t)m * N + n] = val;
      } else {
        int which = n >> 10, d = n & 1023, h = d >> 6, hd = d & 63;
        int b = m >> 11, s = m & 2047;
        _Float16 hv = (_Float16)val;
        size_t bh = (size_t)(b * HH + h);
        if (which == 0)      qh[(bh * SS + s) * HD + hd] = hv;
        else if (which == 1) kh[(bh * SS + s) * HD + hd] = hv;
        else                 vTh[(bh * HD + hd) * SS + s] = hv;  // transposed
      }
    }
  }
}

// ---------------------------------------------------------------------------
// Fused attention: per block = (bh, 32 query rows).
// LDS: scores[32][2052] f32 (256KB, CDNA5-only) + Q tile + reduce scratch.
// Phase B: S = Q K^T / 8 via WMMA into LDS (K-tiles prefetched ahead).
// Phase C: row softmax (8 lanes/row), normalized P -> d_out (single store).
// Phase D: attn_out = P V via WMMA (P: LDS f32->f16; V^T fragments: 16B loads).
// ---------------------------------------------------------------------------
#define SC_STRIDE 2052
#define QT_STRIDE 72
#define ATTN_SMEM_BYTES ((32 * SC_STRIDE) * 4 + 32 * QT_STRIDE * 2 + 32 * 8 * 4 + 32 * 4)

__global__ __launch_bounds__(256) void attn_fused(const _Float16* __restrict__ qh,
                                                  const _Float16* __restrict__ kh,
                                                  const _Float16* __restrict__ vTh,
                                                  float* __restrict__ attnw,
                                                  _Float16* __restrict__ aoh) {
  extern __shared__ float smem[];
  float*    sc    = smem;                                   // [32][2052]
  _Float16* qt    = (_Float16*)(smem + 32 * SC_STRIDE);     // [32][72]
  float*    red   = (float*)(qt + 32 * QT_STRIDE);          // [32][8]
  float*    rstat = red + 32 * 8;                           // [32]

  const int tid  = threadIdx.x;
  const int lane = tid & 31;
  const int w    = tid >> 5;
  const int bh   = blockIdx.y;          // b*16 + h
  const int q0   = blockIdx.x * 32;

  // --- load Q tile (32 x 64 f16) via async DMA ---
  {
    int row = tid >> 3, ch = tid & 7;
    async_copy_b128(qt + row * QT_STRIDE + ch * 8,
                    qh + ((size_t)bh * SS + q0 + row) * HD + ch * 8);
  }
  wait_async_all();
  __syncthreads();

  // --- Phase B: scores = Q K^T * (1/sqrt(64)) ---
  v16h aq[2][2];
#pragma unroll
  for (int mt = 0; mt < 2; ++mt)
#pragma unroll
    for (int kc = 0; kc < 2; ++kc)
      aq[mt][kc] = ld_frag(qt + mt * 16 * QT_STRIDE, QT_STRIDE, kc * 32, lane);

  const int i  = lane & 15;
  const int mo = (lane & 16) ? 8 : 0;
  for (int kt = w; kt < SS / 16; kt += 8) {
    const _Float16* kb = kh + ((size_t)bh * SS + kt * 16) * HD;  // 16 x 64
    if (kt + 8 < SS / 16)  // hide next K tile behind the WMMA chain
      __builtin_prefetch(kb + 8 * 16 * HD, 0, 1);
    v16h b0 = ld_frag(kb, HD, 0, lane);
    v16h b1 = ld_frag(kb, HD, 32, lane);
#pragma unroll
    for (int mt = 0; mt < 2; ++mt) {
      v8f acc = {};
      acc = wmma_f16(aq[mt][0], b0, acc);
      acc = wmma_f16(aq[mt][1], b1, acc);
#pragma unroll
      for (int r = 0; r < 8; ++r)
        sc[(mt * 16 + mo + r) * SC_STRIDE + kt * 16 + i] = acc[r] * 0.125f;
    }
  }
  __syncthreads();

  // --- Phase C: softmax over each row of 2048; write P to global once ---
  {
    const int row = tid >> 3, part = tid & 7;
    float* srow = sc + row * SC_STRIDE;
    const int c0 = part * 256;

    float mx = -1e30f;
    for (int c = 0; c < 256; ++c) mx = fmaxf(mx, srow[c0 + c]);
    red[row * 8 + part] = mx;
    __syncthreads();
    if (part == 0) {
      float m2 = red[row * 8];
      for (int t = 1; t < 8; ++t) m2 = fmaxf(m2, red[row * 8 + t]);
      rstat[row] = m2;
    }
    __syncthreads();
    float rmax = rstat[row];

    float sum = 0.f;
    for (int c = 0; c < 256; ++c) {
      float e = __expf(srow[c0 + c] - rmax);
      srow[c0 + c] = e;
      sum += e;
    }
    __syncthreads();          // all exp writes done before reusing red[]
    red[row * 8 + part] = sum;
    __syncthreads();
    if (part == 0) {
      float s2 = 0.f;
      for (int t = 0; t < 8; ++t) s2 += red[row * 8 + t];
      rstat[row] = s2;
    }
    __syncthreads();
    float inv = 1.0f / rstat[row];

    float4* gw4 = (float4*)(attnw + ((size_t)bh * SS + (q0 + row)) * SS + c0);
    for (int c4 = 0; c4 < 64; ++c4) {
      float4 pv;
      pv.x = srow[c0 + c4 * 4 + 0] * inv;
      pv.y = srow[c0 + c4 * 4 + 1] * inv;
      pv.z = srow[c0 + c4 * 4 + 2] * inv;
      pv.w = srow[c0 + c4 * 4 + 3] * inv;
      srow[c0 + c4 * 4 + 0] = pv.x;
      srow[c0 + c4 * 4 + 1] = pv.y;
      srow[c0 + c4 * 4 + 2] = pv.z;
      srow[c0 + c4 * 4 + 3] = pv.w;
      gw4[c4] = pv;
    }
  }
  __syncthreads();

  // --- Phase D: attn_out(32x64) = P(32x2048) @ V(2048x64) ---
  {
    const int mt = w & 1;           // 16-row half
    const int nt = w >> 1;          // 16-col subtile of HD
    const int ko = (lane & 16) ? 8 : 0;
    const _Float16* vb = vTh + ((size_t)bh * HD + nt * 16) * SS;  // rows = hd
    v8f acc = {};
    for (int kc = 0; kc < SS; kc += 32) {
      v16h af;
      const float* pr = sc + (size_t)(mt * 16 + i) * SC_STRIDE + kc + ko;
#pragma unroll
      for (int t = 0; t < 8; ++t) {
        af[t]     = (_Float16)pr[t];
        af[t + 8] = (_Float16)pr[t + 16];
      }
      v16h bf = ld_frag(vb, SS, kc, lane);
      acc = wmma_f16(af, bf, acc);
    }
    const int b = bh >> 4, h = bh & 15;
#pragma unroll
    for (int r = 0; r < 8; ++r) {
      int s  = q0 + mt * 16 + mo + r;
      int hd = nt * 16 + i;
      aoh[((size_t)(b * SS + s)) * DD + h * HD + hd] = (_Float16)acc[r];
    }
  }
}

// ---------------------------------------------------------------------------
extern "C" void kernel_launch(void* const* d_in, const int* in_sizes, int n_in,
                              void* d_out, int out_size, void* d_ws, size_t ws_size,
                              hipStream_t stream) {
  const float* x      = (const float*)d_in[0];
  const float* w_qkv  = (const float*)d_in[1];
  const float* b_qkv  = (const float*)d_in[2];
  const float* w_proj = (const float*)d_in[3];
  const float* b_proj = (const float*)d_in[4];

  float* out   = (float*)d_out;
  float* attnw = out + (size_t)BB * SS * DD;

  char* ws = (char*)d_ws;
  _Float16* xh     = (_Float16*)ws; ws += (size_t)BB * SS * DD * 2;       // 16 MB
  _Float16* wqkvh  = (_Float16*)ws; ws += (size_t)DD * 3 * DD * 2;        //  6 MB
  _Float16* wprojh = (_Float16*)ws; ws += (size_t)DD * DD * 2;            //  2 MB
  _Float16* qh     = (_Float16*)ws; ws += (size_t)BB * SS * DD * 2;       // 16 MB
  _Float16* kh     = (_Float16*)ws; ws += (size_t)BB * SS * DD * 2;       // 16 MB
  _Float16* vTh    = (_Float16*)ws; ws += (size_t)BB * SS * DD * 2;       // 16 MB
  _Float16* aoh    = (_Float16*)ws; ws += (size_t)BB * SS * DD * 2;       // 16 MB

  // 1) f32 -> f16 conversions
  {
    int n4;
    n4 = (BB * SS * DD) / 4;
    cvt_f32_f16<<<(n4 + 255) / 256, 256, 0, stream>>>(x, xh, n4);
    n4 = (DD * 3 * DD) / 4;
    cvt_f32_f16<<<(n4 + 255) / 256, 256, 0, stream>>>(w_qkv, wqkvh, n4);
    n4 = (DD * DD) / 4;
    cvt_f32_f16<<<(n4 + 255) / 256, 256, 0, stream>>>(w_proj, wprojh, n4);
  }

  // 2) QKV projection: [8192,1024] x [1024,3072] -> Q,K,V^T (f16, head-major)
  gemm16<1><<<dim3(3 * DD / 128, BB * SS / 64), 256, 0, stream>>>(
      xh, wqkvh, b_qkv, nullptr, qh, kh, vTh, BB * SS, 3 * DD, DD);

  // 3) Fused attention (scores + softmax + P@V) with 256KB LDS score tile
  attn_fused<<<dim3(SS / 32, BB * HH), 256, ATTN_SMEM_BYTES, stream>>>(
      qh, kh, vTh, attnw, aoh);

  // 4) Output projection: [8192,1024] x [1024,1024] + bias -> f32 out
  gemm16<0><<<dim3(DD / 128, BB * SS / 64), 256, 0, stream>>>(
      aoh, wprojh, b_proj, out, nullptr, nullptr, nullptr, BB * SS, DD, DD);
}
